// RelaxedWordMoverSimilarity_7980049236097
// MI455X (gfx1250) — compile-verified, matrix operand order checked by reading, not run on hardware
//
#include <hip/hip_runtime.h>
#include <hip/hip_bf16.h>

typedef _Float16 v16h __attribute__((ext_vector_type(16)));
typedef _Float16 h4   __attribute__((ext_vector_type(4)));
typedef float    v8f  __attribute__((ext_vector_type(8)));

constexpr int B  = 128;
constexpr int S1 = 256;
constexpr int S2 = 192;
constexpr int H  = 1024;
constexpr int MBLK = 64;          // rows of x1 per workgroup
constexpr int KC   = 64;          // K-chunk (halves) staged in LDS
constexpr int NK   = H / KC;      // 16 chunks
constexpr int AS   = 72;          // padded LDS row stride (halves) for A/B tiles
constexpr int SSTR = 193;         // padded LDS row stride (floats) for sim tile

// Monotone float<->uint mapping so unsigned atomicMax == float max.
__device__ inline unsigned encf(float f) {
    unsigned b = __float_as_uint(f);
    return (b & 0x80000000u) ? ~b : (b | 0x80000000u);
}
__device__ inline float decf(unsigned e) {
    unsigned b = (e & 0x80000000u) ? (e ^ 0x80000000u) : ~e;
    return __uint_as_float(b);
}

__global__ void rwms_init(unsigned* __restrict__ colmax) {
    int i = blockIdx.x * 256 + threadIdx.x;
    if (i < B * S2) colmax[i] = 0u;   // 0 < encf(-inf): identity for encoded max
}

__global__ __launch_bounds__(256)
void rwms_main(const float* __restrict__ x1, const int* __restrict__ mask1,
               const float* __restrict__ x2, const int* __restrict__ mask2,
               unsigned* __restrict__ colmax, float* __restrict__ sum1p) {
    __shared__ _Float16 Ah[MBLK * AS];   //  9.2 KB  x1 block (f16)
    __shared__ _Float16 Bh[S2 * AS];     // 27.6 KB  x2 block (f16)
    __shared__ float    simS[MBLK * SSTR]; // 49.4 KB normalized masked sim
    __shared__ float    n1i[MBLK], n2i[S2], red[MBLK];
    __shared__ int      m1s[MBLK], m2s[S2];

    const int tid  = threadIdx.x;
    const int lane = tid & 31;
    const int mb   = blockIdx.x;           // 0..3 row block
    const int b    = blockIdx.y;           // batch
    const int wave = tid >> 5;             // 8 waves: 4 (m) x 2 (n)
    const int m0   = (wave & 3) * 16;      // wave's row tile within block
    const int n0   = (wave >> 2) * 96;     // wave's 96-col strip (6 n-tiles)

    const float* x1b = x1 + ((size_t)b * S1 + (size_t)mb * MBLK) * H;
    const float* x2b = x2 + (size_t)b * S2 * H;

    if (tid < MBLK) m1s[tid] = mask1[b * S1 + mb * MBLK + tid];
    if (tid < S2)   m2s[tid] = mask2[b * S2 + tid];

    // ---- inverse norms: 8 lanes per row, 32 rows per pass, 8 passes (256 rows) ----
    {
        const int j = tid & 7, rg = tid >> 3;          // rg in 0..31
        for (int p = 0; p < 8; ++p) {
            const int row = p * 32 + rg;               // 0..63 -> x1, 64..255 -> x2
            const float* src = (row < MBLK) ? (x1b + (size_t)row * H)
                                            : (x2b + (size_t)(row - MBLK) * H);
            float s = 0.f;
            #pragma unroll
            for (int i = 0; i < 32; ++i) {
                float4 f = *reinterpret_cast<const float4*>(src + j * 4 + i * 32);
                s += f.x * f.x + f.y * f.y + f.z * f.z + f.w * f.w;
            }
            s += __shfl_down(s, 4, 8);
            s += __shfl_down(s, 2, 8);
            s += __shfl_down(s, 1, 8);
            if (j == 0) {
                float inv = rsqrtf(fmaxf(s, 1e-16f));
                if (row < MBLK) n1i[row] = inv; else n2i[row - MBLK] = inv;
            }
        }
    }
    __syncthreads();

    // ---- WMMA GEMM: S = x1_blk (64x1024) * x2^T (1024x192), f16 in / f32 acc ----
    v8f acc[6] = {};
    const int r16 = lane & 15;
    const int hi8 = (lane >> 4) * 8;   // K sub-offset per WMMA A/B lane layout

    for (int c = 0; c < NK; ++c) {
        const int k0 = c * KC;
        // stage A: 64x64 f32 -> f16 (4 float4 per thread)
        #pragma unroll
        for (int i = 0; i < 4; ++i) {
            int idx = tid + i * 256, row = idx >> 4, cv = idx & 15;
            float4 f = *reinterpret_cast<const float4*>(x1b + (size_t)row * H + k0 + cv * 4);
            h4 hv; hv.x = (_Float16)f.x; hv.y = (_Float16)f.y;
                   hv.z = (_Float16)f.z; hv.w = (_Float16)f.w;
            *reinterpret_cast<h4*>(&Ah[row * AS + cv * 4]) = hv;
        }
        // stage B: 192x64 f32 -> f16 (12 float4 per thread)
        #pragma unroll
        for (int i = 0; i < 12; ++i) {
            int idx = tid + i * 256, row = idx >> 4, cv = idx & 15;
            float4 f = *reinterpret_cast<const float4*>(x2b + (size_t)row * H + k0 + cv * 4);
            h4 hv; hv.x = (_Float16)f.x; hv.y = (_Float16)f.y;
                   hv.z = (_Float16)f.z; hv.w = (_Float16)f.w;
            *reinterpret_cast<h4*>(&Bh[row * AS + cv * 4]) = hv;
        }
        __syncthreads();

        #pragma unroll
        for (int kk = 0; kk < KC; kk += 32) {
            v16h af;
            {   // A frag: lane row = m0+r16, halves {kk+hi8..+7, kk+16+hi8..+7}
                union { v16h v; float4 q[2]; } u;
                const _Float16* p = &Ah[(m0 + r16) * AS + kk + hi8];
                u.q[0] = *reinterpret_cast<const float4*>(p);
                u.q[1] = *reinterpret_cast<const float4*>(p + 16);
                af = u.v;
            }
            #pragma unroll
            for (int j = 0; j < 6; ++j) {
                v16h bf;
                {   // B frag: lane col = n0+16j+r16 (x2 row), same K pattern
                    union { v16h v; float4 q[2]; } u;
                    const _Float16* p = &Bh[(n0 + j * 16 + r16) * AS + kk + hi8];
                    u.q[0] = *reinterpret_cast<const float4*>(p);
                    u.q[1] = *reinterpret_cast<const float4*>(p + 16);
                    bf = u.v;
                }
                acc[j] = __builtin_amdgcn_wmma_f32_16x16x32_f16(
                    false, af, false, bf, (short)0, acc[j], false, false);
            }
        }
        __syncthreads();
    }

    // ---- normalize + mask into LDS sim tile ----
    {
        const int hi = lane >> 4;
        #pragma unroll
        for (int j = 0; j < 6; ++j) {
            const int col = n0 + j * 16 + r16;
            const float ni2 = n2i[col];
            const int   mm2 = m2s[col];
            #pragma unroll
            for (int v = 0; v < 8; ++v) {
                const int row = m0 + v + hi * 8;   // D layout: VGPR v -> M=v / M=v+8
                float s = acc[j][v] * n1i[row] * ni2;
                simS[row * SSTR + col] = (m1s[row] && mm2) ? s : -__builtin_inff();
            }
        }
    }
    __syncthreads();

    // ---- row max (local, full 192 cols present) ----
    if (tid < MBLK) {
        float r = -__builtin_inff();
        for (int n = 0; n < S2; ++n) r = fmaxf(r, simS[tid * SSTR + n]);
        red[tid] = m1s[tid] ? r : 0.f;
    }
    // ---- column max (partial over 64 rows) -> encoded global atomic max ----
    if (tid < S2 && m2s[tid]) {
        float r = -__builtin_inff();
        for (int m = 0; m < MBLK; ++m) r = fmaxf(r, simS[m * SSTR + tid]);
        atomicMax(&colmax[b * S2 + tid], encf(r));
    }
    __syncthreads();
    if (tid == 0) {
        float s = 0.f;
        for (int i = 0; i < MBLK; ++i) s += red[i];
        sum1p[b * 4 + mb] = s;   // deterministic partial (no float atomics)
    }
}

__global__ __launch_bounds__(256)
void rwms_final(const int* __restrict__ mask1, const int* __restrict__ mask2,
                const unsigned* __restrict__ colmax, const float* __restrict__ sum1p,
                float* __restrict__ out) {
    const int b = blockIdx.x, t = threadIdx.x;
    __shared__ float fs[256];
    __shared__ int   i1[256], i2[256];
    float v2 = 0.f; int c2 = 0;
    if (t < S2) {
        int m = mask2[b * S2 + t];
        if (m) { v2 = decf(colmax[b * S2 + t]); c2 = 1; }
    }
    fs[t] = v2; i2[t] = c2; i1[t] = mask1[b * S1 + t];
    __syncthreads();
    for (int off = 128; off; off >>= 1) {
        if (t < off) { fs[t] += fs[t + off]; i1[t] += i1[t + off]; i2[t] += i2[t + off]; }
        __syncthreads();
    }
    if (t == 0) {
        float s1 = sum1p[b * 4 + 0] + sum1p[b * 4 + 1] + sum1p[b * 4 + 2] + sum1p[b * 4 + 3];
        out[b] = 0.5f * (s1 / (float)i1[0] + fs[0] / (float)i2[0]);
    }
}

extern "C" void kernel_launch(void* const* d_in, const int* in_sizes, int n_in,
                              void* d_out, int out_size, void* d_ws, size_t ws_size,
                              hipStream_t stream) {
    (void)in_sizes; (void)n_in; (void)out_size; (void)ws_size;
    const float* x1    = (const float*)d_in[0];
    const int*   mask1 = (const int*)d_in[1];
    const float* x2    = (const float*)d_in[2];
    const int*   mask2 = (const int*)d_in[3];
    float* out = (float*)d_out;

    unsigned* colmax = (unsigned*)d_ws;                                  // 128*192 u32
    float*    sum1p  = (float*)((char*)d_ws + (size_t)B * S2 * sizeof(unsigned)); // 128*4 f32

    rwms_init<<<(B * S2 + 255) / 256, 256, 0, stream>>>(colmax);
    rwms_main<<<dim3(S1 / MBLK, B), 256, 0, stream>>>(x1, mask1, x2, mask2, colmax, sum1p);
    rwms_final<<<B, 256, 0, stream>>>(mask1, mask2, colmax, sum1p, out);
}
